// update_v_spherenet_48034914238947
// MI455X (gfx1250) — compile-verified, matrix operand order checked by reading, not run on hardware
//
#include <hip/hip_runtime.h>
#include <hip/hip_bf16.h>
#include <cstdint>

// ---------------------------------------------------------------------------
// MI455X notes:
//  - scatter-sum of 512MB edge data dominates (~23us at 23.3 TB/s); use
//    coalesced float4 loads (one wave32 per 128-float edge row) and hardware
//    global_atomic_add_f32 (accumulator is L2-resident: 25.6MB << 192MB L2).
//  - MLP (~23 GFLOP) runs as fused per-wave WMMA: v_wmma_f32_16x16x32_bf16,
//    activations ping-pong through LDS in bf16, f32 accumulation + SiLU.
// ---------------------------------------------------------------------------

typedef __attribute__((ext_vector_type(16))) __bf16 v16bf;
typedef __attribute__((ext_vector_type(8)))  float  v8f;

#define HIDDEN 128
#define OUTE   256
#define WAVES  4
#define ROWS_PER_BLOCK (WAVES * 16)

__device__ __forceinline__ uint16_t f32_to_bf16(float x) {
    return __builtin_bit_cast(uint16_t, (__bf16)x);
}
__device__ __forceinline__ float bf16_to_f32(uint16_t u) {
    return (float)__builtin_bit_cast(__bf16, u);
}

// Build a 16x32 bf16 fragment from a row-major [16+][stride] bf16 buffer.
// CDNA5 16-bit A layout (ISA 7.12.2): lane L (0-15) holds row M=L with
// K = {0..7, 16..23}; lane L+16 holds row M=L with K = {8..15, 24..31};
// each VGPR packs 2 consecutive K values.  Loading W (row-major [N,K]) with
// "row = output column n" produces exactly the B operand for out = act @ W^T.
__device__ __forceinline__ v16bf frag_rowmajor(const uint16_t* base, int stride_elems,
                                               int kblock, int lane) {
    int m  = lane & 15;
    int ko = (lane >> 4) << 3;                     // 0 or 8
    const uint32_t* row = (const uint32_t*)(base + (size_t)m * stride_elems);
    union { v16bf v; uint32_t u[8]; } f;
    int p0 = (kblock + ko) >> 1;                   // dword index of K pair
    int p1 = (kblock + 16 + ko) >> 1;
#pragma unroll
    for (int j = 0; j < 4; ++j) {
        f.u[j]     = row[p0 + j];
        f.u[4 + j] = row[p1 + j];
    }
    return f.v;
}

// One dense layer for a 16-row tile held by this wave:
//   actout[16][256] = op(actin[16][KIN] @ W[256][KIN]^T + bias)
template<int KIN, bool SILU>
__device__ __forceinline__ void layer_wmma(const uint16_t* actin, const uint16_t* W,
                                           const float* bias, uint16_t* actout, int lane) {
    const int n_lane = lane & 15;
    const int mbase  = (lane >> 4) << 3;           // C layout: vgpr r -> M = r + mbase
#pragma unroll 1
    for (int ct = 0; ct < 16; ++ct) {              // 16 column tiles of 16
        v8f c = {};
        const uint16_t* wb = W + (size_t)(ct * 16) * KIN;
#pragma unroll
        for (int kb = 0; kb < KIN; kb += 32) {
            v16bf a = frag_rowmajor(actin, KIN, kb, lane);
            v16bf b = frag_rowmajor(wb,    KIN, kb, lane);
            c = __builtin_amdgcn_wmma_f32_16x16x32_bf16(
                    false, a, false, b, (short)0, c, false, false);
        }
        float bv = bias[ct * 16 + n_lane];
#pragma unroll
        for (int r = 0; r < 8; ++r) {
            float x = c[r] + bv;
            if (SILU) x = x / (1.0f + __expf(-x)); // silu = x*sigmoid(x)
            actout[(mbase + r) * OUTE + ct * 16 + n_lane] = f32_to_bf16(x);
        }
    }
}

// ---------------------------------------------------------------------------

__global__ void __launch_bounds__(256)
zero_f4_kernel(float4* __restrict__ p, int n4) {
    int t = blockIdx.x * blockDim.x + threadIdx.x;
    if (t < n4) p[t] = make_float4(0.f, 0.f, 0.f, 0.f);
}

__global__ void __launch_bounds__(256)
cvt_bf16_kernel(const float* __restrict__ src, uint16_t* __restrict__ dst, int n) {
    int t = blockIdx.x * blockDim.x + threadIdx.x;
    if (t < n) dst[t] = f32_to_bf16(src[t]);
}

// One wave32 per edge: lane reads float4 (512B/row fully coalesced), 4 fp32
// hardware atomics into the L2-resident node accumulator.
__global__ void __launch_bounds__(256)
scatter_kernel(const float* __restrict__ e2, const int* __restrict__ idx,
               float* __restrict__ v, int nEdges) {
    int gid  = blockIdx.x * blockDim.x + threadIdx.x;
    int edge = gid >> 5;
    int lane = gid & 31;
    if (edge >= nEdges) return;
    int node = idx[edge];
    const float4 x = *(const float4*)(e2 + (size_t)edge * HIDDEN + lane * 4);
    float* dst = v + (size_t)node * HIDDEN + lane * 4;
    unsafeAtomicAdd(dst + 0, x.x);
    unsafeAtomicAdd(dst + 1, x.y);
    unsafeAtomicAdd(dst + 2, x.z);
    unsafeAtomicAdd(dst + 3, x.w);
}

// Fused MLP: each wave owns 16 rows; activations ping-pong through LDS bf16.
__global__ void __launch_bounds__(128)
mlp_kernel(const float* __restrict__ vh,
           const uint16_t* __restrict__ Wup, const float* __restrict__ bup,
           const uint16_t* __restrict__ Wl,  const float* __restrict__ bl,
           const uint16_t* __restrict__ Wout,
           float* __restrict__ out, int nNodes) {
    __shared__ __align__(16) uint16_t smem[WAVES][2][16 * OUTE];
    const int wave = threadIdx.x >> 5;
    const int lane = threadIdx.x & 31;
    const int row0 = blockIdx.x * ROWS_PER_BLOCK + wave * 16;

    uint16_t* bufA = smem[wave][0];
    uint16_t* bufB = smem[wave][1];

    // Stage the 16x128 fp32 node tile into LDS as bf16 (OOB rows -> 0).
    for (int e = lane; e < 16 * HIDDEN; e += 32) {
        int r = e >> 7, k = e & (HIDDEN - 1);
        int g = row0 + r;
        float x = (g < nNodes) ? vh[(size_t)g * HIDDEN + k] : 0.0f;
        bufA[r * HIDDEN + k] = f32_to_bf16(x);
    }
    __syncthreads();

    layer_wmma<HIDDEN, false>(bufA, Wup, bup, bufB, lane);            // up-proj
    __syncthreads();
    layer_wmma<OUTE, true>(bufB, Wl + 0 * OUTE * OUTE, bl + 0 * OUTE, bufA, lane);
    __syncthreads();
    layer_wmma<OUTE, true>(bufA, Wl + 1 * OUTE * OUTE, bl + 1 * OUTE, bufB, lane);
    __syncthreads();
    layer_wmma<OUTE, true>(bufB, Wl + 2 * OUTE * OUTE, bl + 2 * OUTE, bufA, lane);
    __syncthreads();

    // Output head (256 -> 1): 2 lanes per row, shuffle-reduce across halves.
    int m = lane & 15, half = lane >> 4;
    const uint16_t* arow = bufA + m * OUTE + half * 128;
    const uint16_t* wrow = Wout + half * 128;
    float s = 0.0f;
#pragma unroll 8
    for (int n = 0; n < 128; ++n)
        s += bf16_to_f32(arow[n]) * bf16_to_f32(wrow[n]);
    s += __shfl_xor(s, 16, 32);
    if (half == 0 && (row0 + m) < nNodes) out[row0 + m] = s;
}

// ---------------------------------------------------------------------------

extern "C" void kernel_launch(void* const* d_in, const int* in_sizes, int n_in,
                              void* d_out, int out_size, void* d_ws, size_t ws_size,
                              hipStream_t stream) {
    (void)n_in; (void)ws_size;
    const float* e2   = (const float*)d_in[0];
    const int*   idx  = (const int*)d_in[1];
    const float* Wup  = (const float*)d_in[2];
    const float* bup  = (const float*)d_in[3];
    const float* Ws   = (const float*)d_in[4];
    const float* bs   = (const float*)d_in[5];
    const float* Wout = (const float*)d_in[6];

    const int nEdges = in_sizes[1];
    const int nNodes = out_size;           // OUT_CH == 1
    float* out = (float*)d_out;

    // Workspace layout (256B aligned sections).
    char* ws = (char*)d_ws;
    float* vh = (float*)ws;
    size_t off = ((size_t)nNodes * HIDDEN * sizeof(float) + 255) & ~(size_t)255;
    uint16_t* Wup_b  = (uint16_t*)(ws + off);
    off += ((size_t)OUTE * HIDDEN * 2 + 255) & ~(size_t)255;
    uint16_t* Ws_b   = (uint16_t*)(ws + off);
    off += ((size_t)3 * OUTE * OUTE * 2 + 255) & ~(size_t)255;
    uint16_t* Wout_b = (uint16_t*)(ws + off);

    // 1) zero the node accumulator
    int n4 = nNodes * HIDDEN / 4;
    zero_f4_kernel<<<(n4 + 255) / 256, 256, 0, stream>>>((float4*)vh, n4);

    // 2) one-shot weight conversion to bf16 (L2-resident afterwards)
    cvt_bf16_kernel<<<(OUTE * HIDDEN + 255) / 256, 256, 0, stream>>>(Wup, Wup_b, OUTE * HIDDEN);
    cvt_bf16_kernel<<<(3 * OUTE * OUTE + 255) / 256, 256, 0, stream>>>(Ws, Ws_b, 3 * OUTE * OUTE);
    cvt_bf16_kernel<<<1, 256, 0, stream>>>(Wout, Wout_b, OUTE);

    // 3) scatter-sum (bandwidth-bound: 512MB edge stream)
    scatter_kernel<<<(nEdges + 7) / 8, 256, 0, stream>>>(e2, idx, vh, nEdges);

    // 4) fused WMMA MLP + output head
    int blocks = (nNodes + ROWS_PER_BLOCK - 1) / ROWS_PER_BLOCK;
    mlp_kernel<<<blocks, 128, 0, stream>>>(vh, Wup_b, bup, Ws_b, bs, Wout_b, out, nNodes);
}
